// TissueMaskExtractor_3539053052638
// MI455X (gfx1250) — matrix operand order repaired and checked by reading, not compiled, not run on hardware
//
#include <hip/hip_runtime.h>

typedef __attribute__((ext_vector_type(16))) _Float16 v16h;
typedef __attribute__((ext_vector_type(8)))  float    v8f;

#define BATCH 64
#define HH 512
#define WW 512
#define NPIX (HH * WW)               // 262144 per image
#define TOTAL ((size_t)BATCH * NPIX) // 16,777,216
#define N_ITERS 32
#define MIN_AREA_I 200

__device__ __forceinline__ int imax(int a, int b) { return a > b ? a : b; }

// ---------------------------------------------------------------------------
// WMMA box-count morphology: one wave computes a 16x16 output tile.
//   T = U(16x32) * P(32x16|16..31)   (vertical band sums, 2 WMMAs)
//   C = T(16x32) * V(32x16)          (horizontal band sums, 1 WMMA)
// U[m][k] = (m<=k<=m+2R), V[k][j] = (j<=k<=j+2R); patch origin (row0-R,col0-R),
// zero-padded ('SAME' conv). Counts <= 25 exact in f16. EXEC all-ones: grid
// covers tiles exactly.
// ---------------------------------------------------------------------------
template<int R, bool IN_IS_FLOAT, bool OUT_IS_FLOAT>
__global__ __launch_bounds__(128) void box_kernel(const void* __restrict__ inv,
                                                  void* __restrict__ outv,
                                                  float inThresh, float cntThresh)
{
    const float*         inF  = (const float*)inv;
    const unsigned char* inB  = (const unsigned char*)inv;
    float*               outF = (float*)outv;
    unsigned char*       outB = (unsigned char*)outv;

    __shared__ _Float16 sP[4][32][34];
    __shared__ float    sT[4][16][34];

    const int wid  = threadIdx.x >> 5;
    const int lane = threadIdx.x & 31;
    const int hf   = lane >> 4;
    const int n    = lane & 15;

    const int tilesX = WW / 16, tilesY = HH / 16;
    const int tile = blockIdx.x * 4 + wid;
    const int img  = tile / (tilesX * tilesY);
    const int trem = tile % (tilesX * tilesY);
    const int row0 = (trem / tilesX) * 16;
    const int col0 = (trem % tilesX) * 16;
    const size_t imgOff = (size_t)img * NPIX;

    for (int idx = lane; idx < 32 * 32; idx += 32) {
        const int pr = idx >> 5, pc = idx & 31;
        const int gr = row0 - R + pr, gc = col0 - R + pc;
        float v = 0.0f;
        if (gr >= 0 && gr < HH && gc >= 0 && gc < WW) {
            if (IN_IS_FLOAT) v = (inF[imgOff + (size_t)gr * WW + gc] > inThresh) ? 1.0f : 0.0f;
            else             v = (float)inB[imgOff + (size_t)gr * WW + gc];
        }
        sP[wid][pr][pc] = (_Float16)v;
    }
    __syncthreads();

    v16h aU;
#pragma unroll
    for (int t = 0; t < 16; ++t) {
        const int v = t >> 1, e = t & 1;
        const int k = ((v >> 2) << 4) + (hf << 3) + ((v & 3) << 1) + e;
        aU[t] = (_Float16)((k >= n && k <= n + 2 * R) ? 1.0f : 0.0f);
    }
    v16h b0, b1;
#pragma unroll
    for (int t = 0; t < 16; ++t) {
        const int v = t >> 1, e = t & 1;
        const int k = ((v >> 2) << 4) + (hf << 3) + ((v & 3) << 1) + e;
        b0[t] = sP[wid][k][n];
        b1[t] = sP[wid][k][16 + n];
    }
    v8f c0 = {}, c1 = {};
    c0 = __builtin_amdgcn_wmma_f32_16x16x32_f16(false, aU, false, b0, (short)0, c0, false, false);
    c1 = __builtin_amdgcn_wmma_f32_16x16x32_f16(false, aU, false, b1, (short)0, c1, false, false);

#pragma unroll
    for (int v = 0; v < 8; ++v) {
        sT[wid][v + 8 * hf][n]      = c0[v];
        sT[wid][v + 8 * hf][16 + n] = c1[v];
    }
    __syncthreads();

    v16h aT, bV;
#pragma unroll
    for (int t = 0; t < 16; ++t) {
        const int v = t >> 1, e = t & 1;
        const int ka = ((v >> 2) << 4) + (hf << 3) + ((v & 3) << 1) + e;
        aT[t] = (_Float16)sT[wid][n][ka];
        const int kb = ((v >> 2) << 4) + (hf << 3) + ((v & 3) << 1) + e;
        bV[t] = (_Float16)((kb >= n && kb <= n + 2 * R) ? 1.0f : 0.0f);
    }
    v8f cnt = {};
    cnt = __builtin_amdgcn_wmma_f32_16x16x32_f16(false, aT, false, bV, (short)0, cnt, false, false);

#pragma unroll
    for (int v = 0; v < 8; ++v) {
        const int m = v + 8 * hf;
        const size_t o = imgOff + (size_t)(row0 + m) * WW + (col0 + n);
        const bool on = cnt[v] > cntThresh;
        if (OUT_IS_FLOAT) outF[o] = on ? 1.0f : 0.0f;
        else              outB[o] = on ? (unsigned char)1 : (unsigned char)0;
    }
}

// ---------------------------------------------------------------------------
// Wave-level segmented max-scan (reset where mask==0), 32 elements per chunk.
// ---------------------------------------------------------------------------
__device__ __forceinline__ int segscan_fwd(int v, bool mk, int lane, int& carry)
{
    int V = mk ? v : 0;
    int F = mk ? 0 : 1;
#pragma unroll
    for (int off = 1; off < 32; off <<= 1) {
        const int pV = __shfl_up(V, off, 32);
        const int pF = __shfl_up(F, off, 32);
        const bool use = lane >= off;
        V = (use && !F) ? imax(V, pV) : V;
        F = use ? (F | pF) : F;
    }
    const int Vc = F ? V : imax(V, carry);
    const int out = mk ? Vc : 0;
    carry = __shfl(out, 31, 32);
    return out;
}

__device__ __forceinline__ int segscan_bwd(int v, bool mk, int lane, int& carry)
{
    int V = mk ? v : 0;
    int F = mk ? 0 : 1;
#pragma unroll
    for (int off = 1; off < 32; off <<= 1) {
        const int pV = __shfl_down(V, off, 32);
        const int pF = __shfl_down(F, off, 32);
        const bool use = lane < 32 - off;
        V = (use && !F) ? imax(V, pV) : V;
        F = use ? (F | pF) : F;
    }
    const int Vc = F ? V : imax(V, carry);
    const int out = mk ? Vc : 0;
    carry = __shfl(out, 0, 32);
    return out;
}

__device__ __forceinline__ int vmax3(const int* __restrict__ S, int r, int j)
{
    int v = S[r * WW + j];
    if (r > 0)      v = imax(v, S[(r - 1) * WW + j]);
    if (r < HH - 1) v = imax(v, S[(r + 1) * WW + j]);
    return v;
}

// fused 3x3 max-pool (8-conn hop) + L->R + R->L run propagation; wave per row
__global__ __launch_bounds__(128) void fused_pool_rowprop(const int* __restrict__ src,
                                                          const unsigned char* __restrict__ m,
                                                          int* __restrict__ dst)
{
    const int wav  = blockIdx.x * 4 + (threadIdx.x >> 5);
    const int lane = threadIdx.x & 31;
    const int b = wav / HH, r = wav % HH;
    const int* S = src + (size_t)b * NPIX;
    const unsigned char* M = m + (size_t)b * NPIX + (size_t)r * WW;
    int* D = dst + (size_t)b * NPIX + (size_t)r * WW;

    int regs[16];
    unsigned maskbits = 0;
    int carry = 0, carryLeft = 0;
    int vmaxCur = vmax3(S, r, lane);
#pragma unroll
    for (int k = 0; k < 16; ++k) {
        const int j = (k << 5) + lane;
        const bool mk = M[j] != 0;
        maskbits |= (mk ? 1u : 0u) << k;
        const int vmaxNext = (k < 15) ? vmax3(S, r, j + 32) : 0;
        int left  = __shfl_up(vmaxCur, 1, 32);
        int right = __shfl_down(vmaxCur, 1, 32);
        const int nfirst = __shfl(vmaxNext, 0, 32);
        left  = (lane == 0)  ? carryLeft : left;   // col-1 (0 at image edge)
        right = (lane == 31) ? nfirst    : right;  // col+1 (0 past image edge)
        int pooled = imax(imax(left, vmaxCur), right);
        pooled = mk ? pooled : 0;
        carryLeft = __shfl(vmaxCur, 31, 32);
        vmaxCur = vmaxNext;
        regs[k] = segscan_fwd(pooled, mk, lane, carry);
    }
    int carryR = 0;
#pragma unroll
    for (int k = 15; k >= 0; --k) {
        const bool mk = (maskbits >> k) & 1u;
        D[(k << 5) + lane] = segscan_bwd(regs[k], mk, lane, carryR);
    }
}

// plain L->R + R->L run propagation (used on transposed field), in place
__global__ __launch_bounds__(128) void rowprop_scan(int* __restrict__ lab,
                                                    const unsigned char* __restrict__ m)
{
    const int wav  = blockIdx.x * 4 + (threadIdx.x >> 5);
    const int lane = threadIdx.x & 31;
    const int b = wav / HH, r = wav % HH;
    int* L = lab + (size_t)b * NPIX + (size_t)r * WW;
    const unsigned char* M = m + (size_t)b * NPIX + (size_t)r * WW;

    int regs[16];
    unsigned maskbits = 0;
    int carry = 0;
#pragma unroll
    for (int k = 0; k < 16; ++k) {
        const int j = (k << 5) + lane;
        const bool mk = M[j] != 0;
        maskbits |= (mk ? 1u : 0u) << k;
        regs[k] = segscan_fwd(L[j], mk, lane, carry);
    }
    int carryR = 0;
#pragma unroll
    for (int k = 15; k >= 0; --k) {
        const bool mk = (maskbits >> k) & 1u;
        L[(k << 5) + lane] = segscan_bwd(regs[k], mk, lane, carryR);
    }
}

// ---------------------------------------------------------------------------
// 32x32 tile transpose; tile staged in LDS via CDNA5 async-to-LDS loads
// (tracked by ASYNCcnt; per-wave s_wait_asynccnt before the block barrier)
// ---------------------------------------------------------------------------
__global__ __launch_bounds__(256) void transpose_i32(const int* __restrict__ src,
                                                     int* __restrict__ dst)
{
    __shared__ int tile[32][33];
    const int b    = blockIdx.x >> 8;
    const int trem = blockIdx.x & 255;
    const int row0 = (trem >> 4) << 5;
    const int col0 = (trem & 15) << 5;
    const int* S = src + (size_t)b * NPIX;
    int*       D = dst + (size_t)b * NPIX;
    const int lx = threadIdx.x & 31;
    const int wy = threadIdx.x >> 5;   // 0..7

#pragma unroll
    for (int i = 0; i < 4; ++i) {
        const int ly = wy + (i << 3);
        const unsigned int ldsoff = (unsigned int)(uintptr_t)&tile[ly][lx];
        const int* g = &S[(size_t)(row0 + ly) * WW + col0 + lx];
        asm volatile("global_load_async_to_lds_b32 %0, %1, off"
                     :: "v"(ldsoff), "v"(g) : "memory");
    }
    asm volatile("s_wait_asynccnt 0x0" ::: "memory");
    __syncthreads();

#pragma unroll
    for (int i = 0; i < 4; ++i) {
        const int ly = wy + (i << 3);
        D[(size_t)(col0 + ly) * WW + row0 + lx] = tile[lx][ly];
    }
}

__global__ __launch_bounds__(256) void transpose_u8(const unsigned char* __restrict__ src,
                                                    unsigned char* __restrict__ dst)
{
    __shared__ unsigned char tile[32][33];
    const int b    = blockIdx.x >> 8;
    const int trem = blockIdx.x & 255;
    const int row0 = (trem >> 4) << 5;
    const int col0 = (trem & 15) << 5;
    const unsigned char* S = src + (size_t)b * NPIX;
    unsigned char*       D = dst + (size_t)b * NPIX;
    const int lx = threadIdx.x & 31;
    const int wy = threadIdx.x >> 5;
#pragma unroll
    for (int i = 0; i < 4; ++i) {
        const int ly = wy + (i << 3);
        tile[ly][lx] = S[(size_t)(row0 + ly) * WW + col0 + lx];
    }
    __syncthreads();
#pragma unroll
    for (int i = 0; i < 4; ++i) {
        const int ly = wy + (i << 3);
        D[(size_t)(col0 + ly) * WW + row0 + lx] = tile[lx][ly];
    }
}

// ---------------------------------------------------------------------------
// seed / histogram / top-2 / keep
// ---------------------------------------------------------------------------
__global__ void seed_kernel(const unsigned char* __restrict__ m, int* __restrict__ lab)
{
    size_t i = (size_t)blockIdx.x * blockDim.x + threadIdx.x;
    if (i >= TOTAL) return;
    const int p = (int)(i % NPIX);
    lab[i] = m[i] ? (p + 1) : 0;
}

__global__ void clear_kernel(int* __restrict__ p, size_t nwords)
{
    size_t i = (size_t)blockIdx.x * blockDim.x + threadIdx.x;
    if (i < nwords) p[i] = 0;
}

__global__ void hist_kernel(const int* __restrict__ lab,
                            const unsigned char* __restrict__ m,
                            int* __restrict__ hist)
{
    size_t i = (size_t)blockIdx.x * blockDim.x + threadIdx.x;
    if (i >= TOTAL) return;
    if (!m[i]) return;
    const int l = lab[i];
    if (l <= 0) return;
    const size_t b = i / NPIX;
    atomicAdd(&hist[b * NPIX + (size_t)(l - 1)], 1);
}

__device__ __forceinline__ void top2_merge(int v, int idx, int& v0, int& i0, int& v1, int& i1)
{
    if (v > v0 || (v == v0 && v > 0 && (unsigned)idx < (unsigned)i0)) {
        v1 = v0; i1 = i0; v0 = v; i0 = idx;
    } else if (v > v1 || (v == v1 && v > 0 && (unsigned)idx < (unsigned)i1)) {
        v1 = v; i1 = idx;
    }
}

__global__ __launch_bounds__(256) void top2_kernel(const int* __restrict__ hist,
                                                   int* __restrict__ top)
{
    const int b = blockIdx.x;
    const int* Hst = hist + (size_t)b * NPIX;
    int v0 = 0, i0 = -1, v1 = 0, i1 = -1;
    for (int k = threadIdx.x; k < NPIX; k += 256) {
        const int a = Hst[k];
        if (a >= MIN_AREA_I) top2_merge(a, k, v0, i0, v1, i1);
    }
    __shared__ int sv0[256], si0[256], sv1[256], si1[256];
    sv0[threadIdx.x] = v0; si0[threadIdx.x] = i0;
    sv1[threadIdx.x] = v1; si1[threadIdx.x] = i1;
    __syncthreads();
    if (threadIdx.x == 0) {
        int V0 = 0, I0 = -1, V1 = 0, I1 = -1;
        for (int t = 0; t < 256; ++t) {
            top2_merge(sv0[t], si0[t], V0, I0, V1, I1);
            top2_merge(sv1[t], si1[t], V0, I0, V1, I1);
        }
        top[b * 2 + 0] = (V0 > 0) ? (I0 + 1) : -1;
        top[b * 2 + 1] = (V1 > 0) ? (I1 + 1) : -1;
    }
}

__global__ void keep_kernel(const int* __restrict__ lab,
                            const unsigned char* __restrict__ m,
                            const int* __restrict__ top,
                            unsigned char* __restrict__ out)
{
    size_t i = (size_t)blockIdx.x * blockDim.x + threadIdx.x;
    if (i >= TOTAL) return;
    const size_t b = i / NPIX;
    const int l = lab[i];
    const int t0 = top[b * 2 + 0], t1 = top[b * 2 + 1];
    out[i] = (m[i] && (l == t0 || l == t1)) ? (unsigned char)1 : (unsigned char)0;
}

// ---------------------------------------------------------------------------
extern "C" void kernel_launch(void* const* d_in, const int* in_sizes, int n_in,
                              void* d_out, int out_size, void* d_ws, size_t ws_size,
                              hipStream_t stream)
{
    (void)in_sizes; (void)n_in; (void)out_size; (void)ws_size;
    const float* x = (const float*)d_in[0];
    float* out = (float*)d_out;

    char* ws = (char*)d_ws;
    const size_t MSZ = TOTAL;
    unsigned char* bufE  = (unsigned char*)ws;                  // 16 MB
    unsigned char* bufM  = (unsigned char*)(ws + MSZ);          // 16 MB (open mask)
    int* lab   = (int*)(ws + 2 * MSZ);                          // 64 MB
    int* labT  = lab + TOTAL;                                   // 64 MB (also hist)
    int* top2  = labT + TOTAL;                                  // 512 B
    unsigned char* maskT = (unsigned char*)(top2 + 256);        // 16 MB

    const int nTileBlocks = (int)(TOTAL / 256 / 4);   // 16384
    const int nPixBlocks  = (int)((TOTAL + 255) / 256);
    const int nRowBlocks  = (BATCH * HH) / 4;         // 8192 (wave per row)
    const int nTrBlocks   = BATCH * 256;              // 16384 (32x32 tiles)

    // 1. threshold + 3x3 erode (==9), 3x3 dilate (>0) -> bufM (opened mask)
    box_kernel<1, true,  false><<<nTileBlocks, 128, 0, stream>>>(x,    bufE, -0.5f, 8.5f);
    box_kernel<1, false, false><<<nTileBlocks, 128, 0, stream>>>(bufE, bufM,  0.0f, 0.5f);
    transpose_u8<<<nTrBlocks, 256, 0, stream>>>(bufM, maskT);

    // 2. CCL: seed, then 32 sweeps of (3x3 hop + row runs) / transpose / col runs
    seed_kernel<<<nPixBlocks, 256, 0, stream>>>(bufM, lab);
    for (int it = 0; it < N_ITERS; ++it) {
        int* src = (it & 1) ? labT : lab;
        int* dst = (it & 1) ? lab  : labT;
        fused_pool_rowprop<<<nRowBlocks, 128, 0, stream>>>(src, bufM, dst);
        transpose_i32<<<nTrBlocks, 256, 0, stream>>>(dst, src);   // src := col-major
        rowprop_scan<<<nRowBlocks, 128, 0, stream>>>(src, maskT); // col runs, in place
        transpose_i32<<<nTrBlocks, 256, 0, stream>>>(src, dst);   // dst := row-major
    }
    // N_ITERS even -> final labels in `lab`; `labT` free, reuse as histogram

    // 3. per-batch area histogram + top-2 components (area >= 200)
    clear_kernel<<<nPixBlocks, 256, 0, stream>>>(labT, TOTAL);
    hist_kernel<<<nPixBlocks, 256, 0, stream>>>(lab, bufM, labT);
    top2_kernel<<<BATCH, 256, 0, stream>>>(labT, top2);
    keep_kernel<<<nPixBlocks, 256, 0, stream>>>(lab, bufM, top2, bufE);

    // 4. hole fill: 5x5 dilate (>0) then 5x5 erode (==25) -> float out
    box_kernel<2, false, false><<<nTileBlocks, 128, 0, stream>>>(bufE, bufM, 0.0f, 0.5f);
    box_kernel<2, false, true ><<<nTileBlocks, 128, 0, stream>>>(bufM, out,  0.0f, 24.5f);
}
